// SparseDistributedMemory_61538291417806
// MI455X (gfx1250) — matrix operand order repaired and checked by reading, not compiled, not run on hardware
//
#include <hip/hip_runtime.h>
#include <stdint.h>

// ---------------------------------------------------------------------------
// Problem constants (match reference setup_inputs)
// ---------------------------------------------------------------------------
constexpr int B_Q    = 2048;    // queries
constexpr int N_ADDR = 65536;   // memory slots
constexpr int A_DIM  = 512;     // address dim
constexpr int D_DIM  = 512;     // data dim
constexpr int TOPK   = 8;
constexpr int NSPLIT = 16;              // N-dimension slices (grid.y of GEMM)
constexpr int NS2    = NSPLIT * 2;      // * 2 column-groups per workgroup
constexpr int MBLK   = 32;              // query rows per workgroup
constexpr int NCOLT  = 64;              // addr columns per tile (2 cg x 32)
constexpr int KCHUNK = 64;              // K staging chunk for B tile (ping-pong)
constexpr int NKC    = A_DIM / KCHUNK;  // 8 chunks
constexpr int BPAD   = 16;              // sB row pad (shorts) = 8 dwords

typedef unsigned short u16;
typedef __attribute__((ext_vector_type(16))) __bf16 v16bf;
typedef __attribute__((ext_vector_type(8)))  float  v8f;
typedef __attribute__((ext_vector_type(4)))  unsigned int v4u;
typedef __attribute__((ext_vector_type(4)))  int    v4i;
typedef __attribute__((ext_vector_type(8)))  int    v8i;

union Frag16 { v16bf v; uint4 q[2]; };

// round-to-nearest-even f32 -> bf16 bit pattern
static __device__ __forceinline__ u16 f2bf(float f) {
    uint32_t u = __float_as_uint(f);
    u += 0x7FFFu + ((u >> 16) & 1u);
    return (u16)(u >> 16);
}

// ---------------------------------------------------------------------------
// Kernel 1: normalize addr rows, convert to bf16.  One wave32 per row.
// ---------------------------------------------------------------------------
__global__ void k_addr_prep(const float* __restrict__ addr, u16* __restrict__ out) {
    const int wave = threadIdx.x >> 5;
    const int lane = threadIdx.x & 31;
    const int row  = blockIdx.x * 8 + wave;            // blockDim=256 -> 8 waves
    const float* src = addr + (size_t)row * A_DIM + lane * 16;

    float4 x0 = ((const float4*)src)[0];
    float4 x1 = ((const float4*)src)[1];
    float4 x2 = ((const float4*)src)[2];
    float4 x3 = ((const float4*)src)[3];

    float s = x0.x*x0.x + x0.y*x0.y + x0.z*x0.z + x0.w*x0.w
            + x1.x*x1.x + x1.y*x1.y + x1.z*x1.z + x1.w*x1.w
            + x2.x*x2.x + x2.y*x2.y + x2.z*x2.z + x2.w*x2.w
            + x3.x*x3.x + x3.y*x3.y + x3.z*x3.z + x3.w*x3.w;
    #pragma unroll
    for (int off = 16; off > 0; off >>= 1) s += __shfl_xor(s, off, 32);

    const float inv = 1.0f / fmaxf(sqrtf(s), 1e-12f);

    u16 o[16];
    const float* xf = &x0.x;
    #pragma unroll
    for (int i = 0; i < 16; ++i) o[i] = f2bf(xf[i] * inv);

    uint4* dst = (uint4*)(out + (size_t)row * A_DIM + lane * 16);
    dst[0] = *(const uint4*)&o[0];
    dst[1] = *(const uint4*)&o[8];
}

// ---------------------------------------------------------------------------
// Kernel 2: q = normalize(query @ W.T + b) -> bf16.  One block per query row.
// ---------------------------------------------------------------------------
__global__ void k_qproj(const float* __restrict__ query, const float* __restrict__ W,
                        const float* __restrict__ bias, u16* __restrict__ qbf) {
    __shared__ __align__(16) float sq[A_DIM];
    __shared__ float so[A_DIM];
    __shared__ float red[256];

    const int b   = blockIdx.x;
    const int tid = threadIdx.x;                       // 256 threads

    for (int i = tid; i < A_DIM; i += 256) sq[i] = query[(size_t)b * A_DIM + i];
    __syncthreads();

    #pragma unroll
    for (int h = 0; h < 2; ++h) {
        const int a = tid + h * 256;
        const float4* wr = (const float4*)(W + (size_t)a * A_DIM);
        float acc = bias[a];
        #pragma unroll 4
        for (int k = 0; k < A_DIM / 4; ++k) {
            float4 w4 = wr[k];
            float4 q4 = ((const float4*)sq)[k];
            acc += w4.x*q4.x + w4.y*q4.y + w4.z*q4.z + w4.w*q4.w;
        }
        so[a] = acc;
    }
    __syncthreads();

    float p = so[tid]*so[tid] + so[tid+256]*so[tid+256];
    red[tid] = p;
    __syncthreads();
    #pragma unroll
    for (int off = 128; off > 0; off >>= 1) {
        if (tid < off) red[tid] += red[tid + off];
        __syncthreads();
    }
    const float inv = 1.0f / fmaxf(sqrtf(red[0]), 1e-12f);

    qbf[(size_t)b * A_DIM + tid]       = f2bf(so[tid]       * inv);
    qbf[(size_t)b * A_DIM + tid + 256] = f2bf(so[tid + 256] * inv);
}

// ---------------------------------------------------------------------------
// TDM issue helper: DMA 64 addr rows x 64 K (bf16, stride 512) into one LDS
// ping-pong buffer.  pad_enable inserts 8 dwords after every 32 dwords ->
// reproduces the padded 160 B sB row stride for bank-conflict-free reads.
// ---------------------------------------------------------------------------
#if __has_builtin(__builtin_amdgcn_tensor_load_to_lds)
static __device__ __forceinline__ void tdm_issue_b(u16 (*dst)[KCHUNK + BPAD],
                                                   const u16* __restrict__ src) {
    const uint64_t ga  = (uint64_t)(uintptr_t)src;
    const unsigned lds = (unsigned)(uintptr_t)&dst[0][0];  // low 32b = LDS offset
    v4u g0 = { 0x1u,                                       // count=1, user D#
               lds,                                        // lds_addr
               (unsigned)ga,                               // global_addr[31:0]
               (unsigned)((ga >> 32) & 0x01FFFFFFu) | 0x80000000u }; // type=2
    v8i g1 = { (int)0x0F110000,      // pad_amt=8dw, pad_int=32dw, pad_en, data_size=2B
               (int)(512u << 16),    // tensor_dim0 = 512 (bits 79:48)
               0,                    // tensor_dim0 hi / tensor_dim1 lo
               (int)0x00400001,      // tensor_dim1 hi (65536>>16) | tile_dim0=64
               64,                   // tile_dim1 = 64 rows
               512,                  // tensor_dim0_stride = 512 elems
               0, 0 };
    v4i z4 = { 0, 0, 0, 0 };
#if __clang_major__ >= 23
    v8i z8 = { 0, 0, 0, 0, 0, 0, 0, 0 };
    __builtin_amdgcn_tensor_load_to_lds(g0, g1, z4, z4, z8, 0);
#else
    __builtin_amdgcn_tensor_load_to_lds(g0, g1, z4, z4, 0);
#endif
}
#else
static __device__ __forceinline__ void coop_load_b(u16 (*dst)[KCHUNK + BPAD],
                                                   const u16* __restrict__ src, int tid) {
    // 64 rows x 8 uint4 per row = 512 uint4 / 128 threads = 4 each
    #pragma unroll
    for (int it = 0; it < NCOLT * (KCHUNK / 8) / 128; ++it) {
        const int i = tid + it * 128;
        const int r = i >> 3, cpos = (i & 7) * 8;
        *(uint4*)&dst[r][cpos] = *(const uint4*)&src[(size_t)r * A_DIM + cpos];
    }
}
#endif

// ---------------------------------------------------------------------------
// Kernel 3: sim = q_bf16 @ addr_bf16^T via v_wmma_f32_16x16x32_bf16.
// B tile is double-buffered in LDS and DMA'd by the Tensor Data Mover, so
// each chunk's transfer overlaps the previous chunk's WMMA work (one barrier
// per chunk; wave 0 hides s_wait_tensorcnt behind the previous compute).
// Fused per-row top-8 over this workgroup's N-slice.
// grid = (B_Q/MBLK, NSPLIT), block = 128 (4 waves = 2 row-grp x 2 col-grp,
// each wave owns a 16x32 output tile = 2 accumulators)
// ---------------------------------------------------------------------------
__global__ void __launch_bounds__(128)
k_sim_topk(const u16* __restrict__ qbf, const u16* __restrict__ abf,
           float* __restrict__ cand_val, int* __restrict__ cand_idx) {
    __shared__ __align__(16) u16 sA[MBLK][A_DIM + 8];          // 33280 B
    __shared__ __align__(16) u16 sB[2][NCOLT][KCHUNK + BPAD];  // 2 x 10240 B
    __shared__ float sC[4][16][17];                            //  4352 B  (=> 58 KB total)

    const int tid  = threadIdx.x;
    const int wave = tid >> 5;
    const int lane = tid & 31;
    const int rg   = wave >> 1;          // row group    (0..1) -> rows rg*16..+15
    const int cg   = wave & 1;           // column group (0..1) -> cols cg*32..+31

    const int mrow0 = blockIdx.x * MBLK;
    const int nbase = blockIdx.y * (N_ADDR / NSPLIT);
    const int ntile = (N_ADDR / NSPLIT) / NCOLT;               // 64 column tiles

    // stage the 32 query rows (full K) once: constant-trip, fully unrolled
    #pragma unroll
    for (int it = 0; it < MBLK * (A_DIM / 8) / 128; ++it) {    // 16 iters
        const int i = tid + it * 128;
        const int r = i >> 6, c = (i & 63) * 8;
        *(uint4*)&sA[r][c] = *(const uint4*)&qbf[(size_t)(mrow0 + r) * A_DIM + c];
    }

    // running top-8 (ascending: tv[0] is the min) for row (mrow0 + rg*16 + lane)
    float tv[TOPK]; int ti[TOPK];
    #pragma unroll
    for (int s = 0; s < TOPK; ++s) { tv[s] = -3.0e38f; ti[s] = 0; }

    const int arow  = rg * 16 + (lane & 15);
    const int ahalf = lane >> 4;
    const int bcol0 = cg * 32 + (lane & 15);                   // sub-tile 0 column
    const int bhalf = lane >> 4;

    for (int ct = 0; ct < ntile; ++ct) {
        const int col0 = nbase + ct * NCOLT;

        // prefetch upcoming tile's address rows (global_prefetch_b8)
        if (ct + 1 < ntile)
            __builtin_prefetch(&abf[(size_t)(col0 + NCOLT) * A_DIM + tid * 64], 0, 1);

        // pipeline prologue: chunk 0 -> buffer 0 (prior readers of buf0 are
        // guaranteed past the previous tile's last barrier)
#if __has_builtin(__builtin_amdgcn_tensor_load_to_lds)
        if (wave == 0) tdm_issue_b(sB[0], &abf[(size_t)col0 * A_DIM]);
#else
        coop_load_b(sB[0], &abf[(size_t)col0 * A_DIM], tid);
#endif

        v8f c0 = {}, c1 = {};
        #pragma unroll 1
        for (int kc = 0; kc < NKC; ++kc) {                     // 8 K-chunks of 64
            const int cur = kc & 1;
#if __has_builtin(__builtin_amdgcn_tensor_load_to_lds)
            if (wave == 0) __builtin_amdgcn_s_wait_tensorcnt(0);  // chunk kc landed
#endif
            __syncthreads();                                   // publish buf[cur]
            if (kc + 1 < NKC) {                                // overlap next DMA
#if __has_builtin(__builtin_amdgcn_tensor_load_to_lds)
                if (wave == 0)
                    tdm_issue_b(sB[cur ^ 1],
                                &abf[(size_t)col0 * A_DIM + (kc + 1) * KCHUNK]);
#else
                coop_load_b(sB[cur ^ 1],
                            &abf[(size_t)col0 * A_DIM + (kc + 1) * KCHUNK], tid);
#endif
            }

            #pragma unroll
            for (int ks = 0; ks < KCHUNK / 32; ++ks) {         // 2 k-steps of 32
                const int kb = kc * KCHUNK + ks * 32;          // global K offset (sA)
                const int kl = ks * 32;                        // chunk-local K (sB)
                Frag16 fa, fb0, fb1;
                fa.q[0]  = *(const uint4*)&sA[arow][kb + ahalf * 16];
                fa.q[1]  = *(const uint4*)&sA[arow][kb + ahalf * 16 + 8];
                fb0.q[0] = *(const uint4*)&sB[cur][bcol0][kl + bhalf * 16];
                fb0.q[1] = *(const uint4*)&sB[cur][bcol0][kl + bhalf * 16 + 8];
                fb1.q[0] = *(const uint4*)&sB[cur][bcol0 + 16][kl + bhalf * 16];
                fb1.q[1] = *(const uint4*)&sB[cur][bcol0 + 16][kl + bhalf * 16 + 8];
                c0 = __builtin_amdgcn_wmma_f32_16x16x32_bf16(
                         false, fa.v, false, fb0.v, (short)0, c0, false, false);
                c1 = __builtin_amdgcn_wmma_f32_16x16x32_bf16(
                         false, fa.v, false, fb1.v, (short)0, c1, false, false);
            }
        }

        // top-8 update: two 16-column phases through per-wave LDS scratch.
        // C layout: vgpr v, lanes 0-15 -> (M=v, N=lane), lanes 16-31 -> (M=v+8).
        const int crow0 = (lane < 16) ? 0 : 8;
        const int ccol  = lane & 15;
        #pragma unroll
        for (int ph = 0; ph < 2; ++ph) {
            #pragma unroll
            for (int v = 0; v < 8; ++v)
                sC[wave][crow0 + v][ccol] = (ph == 0) ? c0[v] : c1[v];
            if (lane < 16) {
                const int r = lane;
                #pragma unroll
                for (int j = 0; j < 16; ++j) {
                    const float v  = sC[wave][r][j];
                    const int gidx = col0 + cg * 32 + ph * 16 + j;
                    if (v > tv[0]) {
                        tv[0] = v; ti[0] = gidx;
                        #pragma unroll
                        for (int s = 0; s < TOPK - 1; ++s) {
                            if (tv[s] > tv[s + 1]) {
                                float fv = tv[s]; tv[s] = tv[s+1]; tv[s+1] = fv;
                                int   fi = ti[s]; ti[s] = ti[s+1]; ti[s+1] = fi;
                            }
                        }
                    }
                }
            }
        }
    }

    if (lane < 16) {
        const int row   = mrow0 + rg * 16 + lane;
        const int slice = blockIdx.y * 2 + cg;
        const size_t base = ((size_t)row * NS2 + slice) * TOPK;
        #pragma unroll
        for (int s = 0; s < TOPK; ++s) { cand_val[base + s] = tv[s]; cand_idx[base + s] = ti[s]; }
    }
}

// ---------------------------------------------------------------------------
// Kernel 4: merge candidates -> global top-8, softmax(x10), gather + blend.
// grid = B_Q, block = 128
// ---------------------------------------------------------------------------
__global__ void k_finalize(const float* __restrict__ cand_val, const int* __restrict__ cand_idx,
                           const float* __restrict__ data_buf, float* __restrict__ out) {
    __shared__ float wts[TOPK];
    __shared__ int   tix[TOPK];

    const int b = blockIdx.x;
    if (threadIdx.x == 0) {
        float tv[TOPK]; int ti[TOPK];
        #pragma unroll
        for (int s = 0; s < TOPK; ++s) { tv[s] = -3.0e38f; ti[s] = 0; }
        const size_t base = (size_t)b * NS2 * TOPK;
        for (int i = 0; i < NS2 * TOPK; ++i) {
            const float v = cand_val[base + i];
            if (v > tv[0]) {
                tv[0] = v; ti[0] = cand_idx[base + i];
                #pragma unroll
                for (int s = 0; s < TOPK - 1; ++s) {
                    if (tv[s] > tv[s + 1]) {
                        float fv = tv[s]; tv[s] = tv[s+1]; tv[s+1] = fv;
                        int   fi = ti[s]; ti[s] = ti[s+1]; ti[s+1] = fi;
                    }
                }
            }
        }
        const float m = tv[TOPK - 1];                  // max (ascending order)
        float w[TOPK], ssum = 0.0f;
        #pragma unroll
        for (int s = 0; s < TOPK; ++s) { w[s] = __expf((tv[s] - m) * 10.0f); ssum += w[s]; }
        const float inv = 1.0f / ssum;
        #pragma unroll
        for (int s = 0; s < TOPK; ++s) { wts[s] = w[s] * inv; tix[s] = ti[s]; }
    }
    __syncthreads();

    const int d = threadIdx.x * 4;                     // 128 threads * float4 = 512
    float4 acc = make_float4(0.f, 0.f, 0.f, 0.f);
    #pragma unroll
    for (int s = 0; s < TOPK; ++s) {
        const float4 dv = *(const float4*)&data_buf[(size_t)tix[s] * D_DIM + d];
        const float  w  = wts[s];
        acc.x += w * dv.x; acc.y += w * dv.y; acc.z += w * dv.z; acc.w += w * dv.w;
    }
    *(float4*)&out[(size_t)b * D_DIM + d] = acc;
}

// ---------------------------------------------------------------------------
// Launch
// ---------------------------------------------------------------------------
extern "C" void kernel_launch(void* const* d_in, const int* /*in_sizes*/, int /*n_in*/,
                              void* d_out, int /*out_size*/, void* d_ws, size_t /*ws_size*/,
                              hipStream_t stream) {
    const float* query = (const float*)d_in[0];
    const float* addr  = (const float*)d_in[1];
    const float* data  = (const float*)d_in[2];
    const float* W     = (const float*)d_in[3];
    const float* bias  = (const float*)d_in[4];
    float* out = (float*)d_out;

    // workspace layout (bytes): addr_bf16 64MB | q_bf16 2MB | cand_val 2MB | cand_idx 2MB
    char* ws = (char*)d_ws;
    u16*   addr_bf16 = (u16*)ws;
    u16*   q_bf16    = (u16*)(ws + (size_t)N_ADDR * A_DIM * 2);
    float* cand_val  = (float*)(ws + (size_t)N_ADDR * A_DIM * 2 + (size_t)B_Q * A_DIM * 2);
    int*   cand_idx  = (int*)((char*)cand_val + (size_t)B_Q * NS2 * TOPK * 4);

    k_addr_prep<<<N_ADDR / 8, 256, 0, stream>>>(addr, addr_bf16);
    k_qproj   <<<B_Q, 256, 0, stream>>>(query, W, bias, q_bf16);
    k_sim_topk<<<dim3(B_Q / MBLK, NSPLIT), 128, 0, stream>>>(q_bf16, addr_bf16, cand_val, cand_idx);
    k_finalize<<<B_Q, 128, 0, stream>>>(cand_val, cand_idx, data, out);
}
